// TransMIL_1219770712645
// MI455X (gfx1250) — compile-verified, matrix operand order checked by reading, not run on hardware
//
#include <hip/hip_runtime.h>
#include <hip/hip_bf16.h>

// ---------------------------------------------------------------------------
// Types for CDNA5 WMMA (wave32) + TDM descriptors
// ---------------------------------------------------------------------------
typedef __attribute__((ext_vector_type(16))) _Float16 v16h;
typedef __attribute__((ext_vector_type(8)))  _Float16 v8h;
typedef __attribute__((ext_vector_type(8)))  float    v8f;
typedef __attribute__((ext_vector_type(4)))  unsigned int u32x4;
typedef __attribute__((ext_vector_type(8)))  int      i32x8;
typedef __attribute__((ext_vector_type(4)))  int      i32x4;

#define LDSK 80   // padded LDS row stride (halves) for a 64-wide K tile (64 + 16 pad)

enum { EF_BIAS = 1, EF_RELU = 2, EF_IDENT = 4 };

// ---------------------------------------------------------------------------
// TDM: DMA one 64-row x 64-half row-major tile into LDS, inserting 16 halves
// (8 dwords) of padding after every 64 halves (32 dwords) so the LDS tile has
// the 80-half padded stride the WMMA fragment loads use.
// D# layout per CDNA5 ISA ch.8 (group0 128b, group1 256b; groups 2/3 unused).
// ---------------------------------------------------------------------------
__device__ __forceinline__ void tdm_tile_to_lds(const _Float16* gsrc, unsigned ldsByteOff,
                                                int ldaElems, int tensorRows) {
  unsigned long long ga = (unsigned long long)(size_t)gsrc;
  u32x4 g0;
  g0.x = 1u;                                              // count=1, user D#
  g0.y = ldsByteOff;                                      // lds_addr (bytes)
  g0.z = (unsigned)ga;                                    // global_addr[31:0]
  g0.w = ((unsigned)(ga >> 32) & 0x01FFFFFFu) | 0x80000000u; // addr[56:32] | type=2
  i32x8 g1;
  g1[0] = (1 << 16)            // data_size = 1 -> 2 bytes
        | (1 << 20)            // pad_enable
        | (4 << 22)            // pad_interval: 32 dwords (= 64 halves)
        | (7 << 25);           // pad_amount:   8 dwords  (= 16 halves)
  g1[1] = (ldaElems & 0xFFFF) << 16;                         // tensor_dim0[15:0]
  g1[2] = ((ldaElems >> 16) & 0xFFFF) | ((tensorRows & 0xFFFF) << 16);
  g1[3] = ((tensorRows >> 16) & 0xFFFF) | (64 << 16);        // tile_dim0 = 64
  g1[4] = 64;                                                // tile_dim1 = 64, tile_dim2 = 0
  g1[5] = ldaElems;                                          // tensor_dim0_stride lo
  g1[6] = 0;                                                 // stride hi / dim1_stride lo
  g1[7] = 0;
  i32x4 gz4 = (i32x4){0, 0, 0, 0};
#if __clang_major__ >= 23
  i32x8 gz8 = (i32x8){0, 0, 0, 0, 0, 0, 0, 0};
  __builtin_amdgcn_tensor_load_to_lds(g0, g1, gz4, gz4, gz8, 0);
#else
  __builtin_amdgcn_tensor_load_to_lds(g0, g1, gz4, gz4, 0);
#endif
}

// Build a 16x32 f16 fragment from an LDS tile stored [row][k] (stride LDSK).
// ISA 7.12.2: lanes 0-15 hold M=row with K={kOff..+7, kOff+16..+23};
// lanes 16-31 shift K by 8. B mirrors with N on lanes.
__device__ __forceinline__ v16h frag_from_lds(const _Float16* base, int row, int kOff, int lane) {
  const _Float16* p = base + row * LDSK + kOff + (lane >> 4) * 8;
  v8h lo = *(const v8h*)(p);        // ds_load_b128
  v8h hi = *(const v8h*)(p + 16);   // ds_load_b128
  v16h r;
#pragma unroll
  for (int i = 0; i < 8; ++i) { r[i] = lo[i]; r[i + 8] = hi[i]; }
  return r;
}

// ---------------------------------------------------------------------------
// Generic batched GEMM, f16 operands, f32 WMMA accumulate, f32 and/or f16 out.
//   B_IS_KN == 0 : C[m,n] = sum_k A[m,k]*B[n,k]  (B row-major (N,K), "NT")
//   B_IS_KN == 1 : C[m,n] = sum_k A[m,k]*B[k,n]  (B row-major (K,N), "NN")
// Epilogue: C = alpha*(A*B) [+ beta*I] [+ bias[n]] [ReLU];  store f32 / f16.
// Block: 64x64 C tile, 128 thr = 4 wave32, each wave 32x32 (2x2 WMMA tiles),
// K-step 64 (two 16x16x32 sub-steps). Requires M%64==0, N%64==0, K%64==0.
//
// Double-buffered pipeline: TDM (wave0 = A panel, wave1 = NT B panel) DMAs
// tile t+1 into the spare LDS buffer while all waves run WMMA on tile t; the
// issuing wave drains TENSORcnt just before the single per-tile barrier, so
// the DMA latency is hidden behind the matrix math. NN-B (true transpose) is
// staged manually into the spare buffer during the same window.
// ---------------------------------------------------------------------------
template <int B_IS_KN>
__global__ __launch_bounds__(128) void gemm_wmma_h(
    const _Float16* __restrict__ Ag, const _Float16* __restrict__ Bg,
    const float* __restrict__ bias,
    float* __restrict__ Cf, _Float16* __restrict__ Ch,
    int M, int N, int K, int lda, int ldb, int ldc,
    long sA, long sB, long sC,
    float alpha, float beta, int flags)
{
  __shared__ __align__(16) _Float16 As[2][64 * LDSK];
  __shared__ __align__(16) _Float16 Bs[2][64 * LDSK];

  const int tid  = threadIdx.x;
  const int lane = tid & 31;
  const int wave = tid >> 5;
  const int wrow = wave >> 1;       // 0..1
  const int wcol = wave & 1;        // 0..1
  const int bm   = blockIdx.y * 64;
  const int bn   = blockIdx.x * 64;

  const _Float16* A = Ag + (size_t)blockIdx.z * (size_t)sA;
  const _Float16* B = Bg + (size_t)blockIdx.z * (size_t)sB;

  v8f acc[2][2];
  const v8f vzero = {0.f,0.f,0.f,0.f,0.f,0.f,0.f,0.f};
#pragma unroll
  for (int i = 0; i < 2; ++i)
#pragma unroll
    for (int j = 0; j < 2; ++j) acc[i][j] = vzero;

  const unsigned asOff[2] = { (unsigned)(size_t)(&As[0][0]), (unsigned)(size_t)(&As[1][0]) };
  const unsigned bsOff[2] = { (unsigned)(size_t)(&Bs[0][0]), (unsigned)(size_t)(&Bs[1][0]) };

  // manual transpose staging for NN-B: Bs[buf][n][k] = B[(kt+k)*ldb + bn+n]
  auto stageB = [&](int kt, int buf) {
    const int n  = tid >> 1;
    const int k0 = (tid & 1) * 32;
    const _Float16* bp = B + (size_t)(kt + k0) * ldb + (bn + n);
    if (kt + 128 < K) __builtin_prefetch(bp + (size_t)128 * ldb, 0, 1);
#pragma unroll 8
    for (int i = 0; i < 32; ++i)
      Bs[buf][n * LDSK + k0 + i] = bp[i * ldb];
  };

  const int nkt = K >> 6;

  // ---- prologue: stage tile 0 into buffer 0 ----
  if (wave == 0) tdm_tile_to_lds(A + (size_t)bm * lda, asOff[0], lda, M);
  if (B_IS_KN == 0) {
    if (wave == 1) tdm_tile_to_lds(B + (size_t)bn * ldb, bsOff[0], ldb, N);
  } else {
    stageB(0, 0);
  }
  if (wave == 0 || (B_IS_KN == 0 && wave == 1)) __builtin_amdgcn_s_wait_tensorcnt(0);
  __syncthreads();

  int cur = 0;
  for (int t = 0; t < nkt; ++t) {
    const int nxt = cur ^ 1;
    const bool hasNext = (t + 1 < nkt);
    const int ktn = (t + 1) << 6;

    // kick off DMA for tile t+1 into the spare buffer (freed by last barrier)
    if (hasNext && wave == 0)
      tdm_tile_to_lds(A + (size_t)bm * lda + ktn, asOff[nxt], lda, M);
    if (B_IS_KN == 0 && hasNext && wave == 1)
      tdm_tile_to_lds(B + (size_t)bn * ldb + ktn, bsOff[nxt], ldb, N);

    // compute on tile t while the DMA streams tile t+1
#pragma unroll
    for (int ks = 0; ks < 64; ks += 32) {
      v16h af[2], bf[2];
#pragma unroll
      for (int i = 0; i < 2; ++i)
        af[i] = frag_from_lds(&As[cur][0], wrow * 32 + i * 16 + (lane & 15), ks, lane);
#pragma unroll
      for (int j = 0; j < 2; ++j)
        bf[j] = frag_from_lds(&Bs[cur][0], wcol * 32 + j * 16 + (lane & 15), ks, lane);
#pragma unroll
      for (int i = 0; i < 2; ++i)
#pragma unroll
        for (int j = 0; j < 2; ++j)
          acc[i][j] = __builtin_amdgcn_wmma_f32_16x16x32_f16(
              false, af[i], false, bf[j], (short)0, acc[i][j], false, false);
    }

    // NN-B: stage the next transposed panel into the spare buffer
    if (B_IS_KN == 1 && hasNext) stageB(ktn, nxt);

    // drain the in-flight DMA (issuing wave only), then one barrier per tile
    if (hasNext && (wave == 0 || (B_IS_KN == 0 && wave == 1)))
      __builtin_amdgcn_s_wait_tensorcnt(0);
    __syncthreads();
    cur = nxt;
  }

  // Epilogue. C/D layout: VGPR r -> M = r + 8*(lane>>4), N = lane&15.
  const int mh = (lane >> 4) * 8;
  const int nl = lane & 15;
  float* CfB = Cf ? Cf + (size_t)blockIdx.z * (size_t)sC : nullptr;
  _Float16* ChB = Ch ? Ch + (size_t)blockIdx.z * (size_t)sC : nullptr;
#pragma unroll
  for (int i = 0; i < 2; ++i) {
#pragma unroll
    for (int j = 0; j < 2; ++j) {
      const int ng = bn + wcol * 32 + j * 16 + nl;
      const float bv = (flags & EF_BIAS) ? bias[ng] : 0.0f;
#pragma unroll
      for (int r = 0; r < 8; ++r) {
        const int mg = bm + wrow * 32 + i * 16 + mh + r;
        float val = alpha * acc[i][j][r] + bv;
        if (flags & EF_IDENT) val += (mg == ng) ? beta : 0.0f;
        if (flags & EF_RELU)  val = fmaxf(val, 0.0f);
        const size_t off = (size_t)mg * ldc + ng;
        if (CfB) CfB[off] = val;
        if (ChB) ChB[off] = (_Float16)val;
      }
    }
  }
}

// ---------------------------------------------------------------------------
// Elementwise / reduction kernels
// ---------------------------------------------------------------------------

__global__ void f32_to_f16(const float* __restrict__ s, _Float16* __restrict__ d, int n) {
  for (int i = blockIdx.x * blockDim.x + threadIdx.x; i < n; i += gridDim.x * blockDim.x)
    d[i] = (_Float16)s[i];
}

// h[b,0,:]=cls ; h[b,1+t,:]=feat0[b*4096+t,:]
__global__ void assemble_h(const float* __restrict__ feat0, const float* __restrict__ cls,
                           float* __restrict__ h, int total) {
  for (int i = blockIdx.x * blockDim.x + threadIdx.x; i < total; i += gridDim.x * blockDim.x) {
    int c = i % 512; int t = (i / 512) % 4097; int b = i / (512 * 4097);
    h[i] = (t == 0) ? cls[c] : feat0[((size_t)b * 4096 + (t - 1)) * 512 + c];
  }
}

// LayerNorm of h (B,4097,512) -> f16 xn (B,4352,512), 255 leading zero rows/batch.
__global__ __launch_bounds__(128) void layernorm_pad(
    const float* __restrict__ h, const float* __restrict__ g,
    const float* __restrict__ bta, _Float16* __restrict__ xn) {
  const int row = blockIdx.x;             // 0 .. 4*4352-1
  const int b = row / 4352, t = row % 4352;
  _Float16* out = xn + (size_t)row * 512;
  const int tid = threadIdx.x;
  if (t < 255) { for (int c = tid; c < 512; c += 128) out[c] = (_Float16)0.0f; return; }
  const float* x = h + ((size_t)b * 4097 + (t - 255)) * 512;
  __shared__ float red[128];
  float s = 0.f;
  for (int c = tid; c < 512; c += 128) s += x[c];
  red[tid] = s; __syncthreads();
  for (int o = 64; o; o >>= 1) { if (tid < o) red[tid] += red[tid + o]; __syncthreads(); }
  const float mu = red[0] / 512.0f; __syncthreads();
  float vs = 0.f;
  for (int c = tid; c < 512; c += 128) { float d = x[c] - mu; vs += d * d; }
  red[tid] = vs; __syncthreads();
  for (int o = 64; o; o >>= 1) { if (tid < o) red[tid] += red[tid + o]; __syncthreads(); }
  const float inv = rsqrtf(red[0] / 512.0f + 1e-5f);
  for (int c = tid; c < 512; c += 128) out[c] = (_Float16)((x[c] - mu) * inv * g[c] + bta[c]);
}

// Split f16 qkv (B,4352,1536) -> f16 q,k,v (B*H,4352,64), q scaled by DH^-0.5.
__global__ void qkv_repack(const _Float16* __restrict__ qkv, _Float16* __restrict__ q,
                           _Float16* __restrict__ k, _Float16* __restrict__ v, int total) {
  for (int i = blockIdx.x * blockDim.x + threadIdx.x; i < total; i += gridDim.x * blockDim.x) {
    int d = i & 63; int n = (i >> 6) % 4352; int g = i / (4352 * 64);
    int b = g >> 3, hh = g & 7;
    size_t src = ((size_t)(b * 4352 + n)) * 1536 + hh * 64 + d;
    q[i] = (_Float16)((float)qkv[src] * 0.125f);
    k[i] = qkv[src + 512];
    v[i] = qkv[src + 1024];
  }
}

// Landmarks: mean over l=17 contiguous rows (f32 accumulate).
__global__ void landmark_mean(const _Float16* __restrict__ q, const _Float16* __restrict__ k,
                              _Float16* __restrict__ ql, _Float16* __restrict__ kl, int total) {
  for (int i = blockIdx.x * blockDim.x + threadIdx.x; i < total; i += gridDim.x * blockDim.x) {
    int d = i & 63; int m = (i >> 6) & 255; int g = i >> 14;
    size_t base = (size_t)g * 4352 * 64 + (size_t)m * 17 * 64 + d;
    float sq = 0.f, sk = 0.f;
#pragma unroll 17
    for (int j = 0; j < 17; ++j) { sq += (float)q[base + j * 64]; sk += (float)k[base + j * 64]; }
    ql[i] = (_Float16)(sq * (1.0f / 17.0f));
    kl[i] = (_Float16)(sk * (1.0f / 17.0f));
  }
}

// Row softmax on f16 storage, f32 math. One block per row.
__global__ __launch_bounds__(256) void softmax_rows_h(_Float16* __restrict__ X, int cols) {
  _Float16* x = X + (size_t)blockIdx.x * cols;
  const int tid = threadIdx.x;
  __shared__ float r[256];
  float m = -3.4e38f;
  for (int c = tid; c < cols; c += 256) m = fmaxf(m, (float)x[c]);
  r[tid] = m; __syncthreads();
  for (int o = 128; o; o >>= 1) { if (tid < o) r[tid] = fmaxf(r[tid], r[tid + o]); __syncthreads(); }
  m = r[0]; __syncthreads();
  float s = 0.f;
  for (int c = tid; c < cols; c += 256) s += __expf((float)x[c] - m);
  r[tid] = s; __syncthreads();
  for (int o = 128; o; o >>= 1) { if (tid < o) r[tid] += r[tid + o]; __syncthreads(); }
  const float inv = 1.0f / r[0];
  for (int c = tid; c < cols; c += 256) x[c] = (_Float16)(__expf((float)x[c] - m) * inv);
}

// Pinv init: per (g,i) row-sum and col-sum of |a2| (32 matrices of 256x256).
__global__ __launch_bounds__(256) void pinv_sums(const _Float16* __restrict__ a, float* __restrict__ red) {
  const int g = blockIdx.x, i = threadIdx.x;
  const _Float16* mat = a + ((size_t)g << 16);
  float rs = 0.f, cs = 0.f;
  for (int t = 0; t < 256; ++t) { rs += fabsf((float)mat[i * 256 + t]); cs += fabsf((float)mat[t * 256 + i]); }
  red[g * 256 + i] = rs;
  red[8192 + g * 256 + i] = cs;
}

// red[16384] = 1 / (max(rowsums) * max(colsums)) -- global max like the reference.
__global__ __launch_bounds__(256) void pinv_finish(float* __restrict__ red) {
  const int tid = threadIdx.x;
  float rm = 0.f, cm = 0.f;
  for (int i = tid; i < 8192; i += 256) { rm = fmaxf(rm, red[i]); cm = fmaxf(cm, red[8192 + i]); }
  __shared__ float r1[256], r2[256];
  r1[tid] = rm; r2[tid] = cm; __syncthreads();
  for (int o = 128; o; o >>= 1) {
    if (tid < o) { r1[tid] = fmaxf(r1[tid], r1[tid + o]); r2[tid] = fmaxf(r2[tid], r2[tid + o]); }
    __syncthreads();
  }
  if (tid == 0) red[16384] = 1.0f / (r1[0] * r2[0]);
}

// z0 = a2^T * (1/denom)   (f16 in/out)
__global__ void transpose_scale(_Float16* __restrict__ z, const _Float16* __restrict__ a,
                                const float* __restrict__ scal, int total) {
  const float s = scal[0];
  for (int i = blockIdx.x * blockDim.x + threadIdx.x; i < total; i += gridDim.x * blockDim.x) {
    int j = i & 255; int r = (i >> 8) & 255; int g = i >> 16;
    z[i] = (_Float16)((float)a[((size_t)g << 16) + (size_t)j * 256 + r] * s);
  }
}

// dst = beta*I - src (f16, 32 batches of 256x256)
__global__ void ident_minus(_Float16* __restrict__ dst, const _Float16* __restrict__ src,
                            float beta, int total) {
  for (int i = blockIdx.x * blockDim.x + threadIdx.x; i < total; i += gridDim.x * blockDim.x) {
    int j = i & 255; int r = (i >> 8) & 255;
    dst[i] = (_Float16)(((r == j) ? beta : 0.0f) - (float)src[i]);
  }
}

// core(f32) += depthwise 33x1 conv of v(f16) along token axis (per head).
__global__ void res_conv_add(float* __restrict__ core, const _Float16* __restrict__ v,
                             const float* __restrict__ w, int total) {
  for (int i = blockIdx.x * blockDim.x + threadIdx.x; i < total; i += gridDim.x * blockDim.x) {
    int d = i & 63; int n = (i >> 6) % 4352; int g = i / (4352 * 64);
    int hh = g & 7;
    float a = 0.f;
#pragma unroll 33
    for (int t = 0; t < 33; ++t) {
      int nn = n + t - 16;
      if (nn >= 0 && nn < 4352)
        a += (float)v[(size_t)g * 4352 * 64 + (size_t)nn * 64 + d] * w[hh * 33 + t];
    }
    core[i] += a;
  }
}

// (B*H,4352,64) f32 -> (B,4352,512) f16
__global__ void merge_heads(const float* __restrict__ core, _Float16* __restrict__ m, int total) {
  for (int i = blockIdx.x * blockDim.x + threadIdx.x; i < total; i += gridDim.x * blockDim.x) {
    int c = i & 511; int n = (i >> 9) % 4352; int b = i / (4352 * 512);
    int hh = c >> 6, d = c & 63;
    m[i] = (_Float16)core[((size_t)(b * 8 + hh) * 4352 + n) * 64 + d];
  }
}

// h (B,4097,512) += attn[:, 255:, :] (B,4352,512)
__global__ void residual_add(float* __restrict__ h, const float* __restrict__ attn, int total) {
  for (int i = blockIdx.x * blockDim.x + threadIdx.x; i < total; i += gridDim.x * blockDim.x) {
    int c = i % 512; int t = (i / 512) % 4097; int b = i / (512 * 4097);
    h[i] += attn[((size_t)b * 4352 + 255 + t) * 512 + c];
  }
}

// Fused identity + 7x7 + 5x5 + 3x3 depthwise convs on the 64x64 token grid,
// in-place on h tokens 1..4096. One block per (b, c), 70x70 halo plane in LDS.
__global__ __launch_bounds__(256) void dwconv_fused(
    float* __restrict__ h,
    const float* __restrict__ w7, const float* __restrict__ b7,
    const float* __restrict__ w5, const float* __restrict__ b5,
    const float* __restrict__ w3, const float* __restrict__ b3) {
  const int b = blockIdx.x / 512;
  const int c = blockIdx.x % 512;
  const int tid = threadIdx.x;
  __shared__ float plane[70][70];
  __shared__ float wk[83];
  if (tid < 49)      wk[tid] = w7[c * 49 + tid];
  else if (tid < 74) wk[tid] = w5[c * 25 + (tid - 49)];
  else if (tid < 83) wk[tid] = w3[c * 9 + (tid - 74)];
  for (int idx = tid; idx < 70 * 70; idx += 256) {
    int yy = idx / 70 - 3, xx = idx % 70 - 3;
    float val = 0.f;
    if (yy >= 0 && yy < 64 && xx >= 0 && xx < 64)
      val = h[((size_t)b * 4097 + 1 + yy * 64 + xx) * 512 + c];
    plane[idx / 70][idx % 70] = val;
  }
  __syncthreads();
  const float bias = b7[c] + b5[c] + b3[c];
  for (int pix = tid; pix < 4096; pix += 256) {
    int y = pix >> 6, x = pix & 63;
    float acc = plane[y + 3][x + 3] + bias;      // identity path
#pragma unroll
    for (int dy = 0; dy < 7; ++dy)
#pragma unroll
      for (int dx = 0; dx < 7; ++dx) acc += plane[y + dy][x + dx] * wk[dy * 7 + dx];
#pragma unroll
    for (int dy = 0; dy < 5; ++dy)
#pragma unroll
      for (int dx = 0; dx < 5; ++dx) acc += plane[y + 1 + dy][x + 1 + dx] * wk[49 + dy * 5 + dx];
#pragma unroll
    for (int dy = 0; dy < 3; ++dy)
#pragma unroll
      for (int dx = 0; dx < 3; ++dx) acc += plane[y + 2 + dy][x + 2 + dx] * wk[74 + dy * 3 + dx];
    h[((size_t)b * 4097 + 1 + pix) * 512 + c] = acc;
  }
}

// Final: LN of cls row, 2-way logits, softmax, argmax. out = [logits8|probs8|yhat4].
__global__ __launch_bounds__(128) void final_head(
    const float* __restrict__ h, const float* __restrict__ g, const float* __restrict__ bt,
    const float* __restrict__ W2, const float* __restrict__ b2, float* __restrict__ out) {
  const int b = blockIdx.x, tid = threadIdx.x;
  const float* x = h + (size_t)b * 4097 * 512;
  __shared__ float r1[128], r2[128];
  float s = 0.f;
  for (int c = tid; c < 512; c += 128) s += x[c];
  r1[tid] = s; __syncthreads();
  for (int o = 64; o; o >>= 1) { if (tid < o) r1[tid] += r1[tid + o]; __syncthreads(); }
  const float mu = r1[0] / 512.0f; __syncthreads();
  float vs = 0.f;
  for (int c = tid; c < 512; c += 128) { float d = x[c] - mu; vs += d * d; }
  r1[tid] = vs; __syncthreads();
  for (int o = 64; o; o >>= 1) { if (tid < o) r1[tid] += r1[tid + o]; __syncthreads(); }
  const float inv = rsqrtf(r1[0] / 512.0f + 1e-5f); __syncthreads();
  float p0 = 0.f, p1 = 0.f;
  for (int c = tid; c < 512; c += 128) {
    float ln = (x[c] - mu) * inv * g[c] + bt[c];
    p0 += ln * W2[c]; p1 += ln * W2[512 + c];
  }
  r1[tid] = p0; r2[tid] = p1; __syncthreads();
  for (int o = 64; o; o >>= 1) {
    if (tid < o) { r1[tid] += r1[tid + o]; r2[tid] += r2[tid + o]; }
    __syncthreads();
  }
  if (tid == 0) {
    const float l0 = r1[0] + b2[0], l1 = r2[0] + b2[1];
    const float mx = fmaxf(l0, l1);
    const float e0 = __expf(l0 - mx), e1 = __expf(l1 - mx);
    const float si = 1.0f / (e0 + e1);
    out[b * 2 + 0] = l0;            out[b * 2 + 1] = l1;
    out[8 + b * 2 + 0] = e0 * si;   out[8 + b * 2 + 1] = e1 * si;
    out[16 + b] = (l1 > l0) ? 1.0f : 0.0f;
  }
}

// ---------------------------------------------------------------------------
// Host orchestration
// ---------------------------------------------------------------------------
extern "C" void kernel_launch(void* const* d_in, const int* in_sizes, int n_in,
                              void* d_out, int out_size, void* d_ws, size_t ws_size,
                              hipStream_t stream) {
  (void)in_sizes; (void)n_in; (void)out_size; (void)ws_size;

  const float* data   = (const float*)d_in[0];
  const float* W1     = (const float*)d_in[1];
  const float* b1     = (const float*)d_in[2];
  const float* cls    = (const float*)d_in[3];
  const float* ln1_g  = (const float*)d_in[4];
  const float* ln1_b  = (const float*)d_in[5];
  const float* Wqkv1  = (const float*)d_in[6];
  const float* Wo1    = (const float*)d_in[7];
  const float* bo1    = (const float*)d_in[8];
  const float* res1   = (const float*)d_in[9];
  const float* ln2_g  = (const float*)d_in[10];
  const float* ln2_b  = (const float*)d_in[11];
  const float* Wqkv2  = (const float*)d_in[12];
  const float* Wo2    = (const float*)d_in[13];
  const float* bo2    = (const float*)d_in[14];
  const float* res2   = (const float*)d_in[15];
  const float* pw7    = (const float*)d_in[16];
  const float* pb7    = (const float*)d_in[17];
  const float* pw5    = (const float*)d_in[18];
  const float* pb5    = (const float*)d_in[19];
  const float* pw3    = (const float*)d_in[20];
  const float* pb3    = (const float*)d_in[21];
  const float* lnf_g  = (const float*)d_in[22];
  const float* lnf_b  = (const float*)d_in[23];
  const float* W2     = (const float*)d_in[24];
  const float* b2     = (const float*)d_in[25];

  // ---- workspace carve ----
  char* wp = (char*)d_ws;
  auto allocF = [&](size_t e) { float* r = (float*)wp;    wp += ((e * 4) + 255) & ~(size_t)255; return r; };
  auto allocH = [&](size_t e) { _Float16* r = (_Float16*)wp; wp += ((e * 2) + 255) & ~(size_t)255; return r; };
  // f32
  float* feat0  = allocF((size_t)16384 * 512);
  float* hbuf   = allocF((size_t)4 * 4097 * 512);
  float* core   = allocF((size_t)32 * 4352 * 64);
  float* attn   = allocF((size_t)4 * 4352 * 512);
  float* red    = allocF(16384 + 8);
  // f16
  _Float16* data_h   = allocH((size_t)4 * 4096 * 1536);
  _Float16* W1_h     = allocH((size_t)512 * 1536);
  _Float16* Wqkv1_h  = allocH((size_t)1536 * 512);
  _Float16* Wqkv2_h  = allocH((size_t)1536 * 512);
  _Float16* Wo1_h    = allocH((size_t)512 * 512);
  _Float16* Wo2_h    = allocH((size_t)512 * 512);
  _Float16* xn_h     = allocH((size_t)4 * 4352 * 512);
  _Float16* qkv_h    = allocH((size_t)4 * 4352 * 1536);
  _Float16* qb       = allocH((size_t)32 * 4352 * 64);
  _Float16* kb       = allocH((size_t)32 * 4352 * 64);
  _Float16* vb       = allocH((size_t)32 * 4352 * 64);
  _Float16* qland    = allocH((size_t)32 * 256 * 64);
  _Float16* kland    = allocH((size_t)32 * 256 * 64);
  _Float16* s1h      = allocH((size_t)32 * 4352 * 256);
  _Float16* s2h      = allocH((size_t)32 * 256 * 256);
  _Float16* s3h      = allocH((size_t)32 * 256 * 4352);
  _Float16* z0h      = allocH((size_t)32 * 256 * 256);
  _Float16* z1h      = allocH((size_t)32 * 256 * 256);
  _Float16* xzh      = allocH((size_t)32 * 256 * 256);
  _Float16* t1h      = allocH((size_t)32 * 256 * 256);
  _Float16* t2h      = allocH((size_t)32 * 256 * 256);
  _Float16* w1h      = allocH((size_t)32 * 256 * 64);
  _Float16* w2h      = allocH((size_t)32 * 256 * 64);
  _Float16* merged_h = allocH((size_t)4 * 4352 * 512);

  auto gsz = [](int n) { return (unsigned)((n + 255) / 256); };
  auto launch_gemm = [&](bool bkn, const _Float16* A, const _Float16* Bm, const float* bias,
                         float* Cf, _Float16* Ch,
                         int M, int N, int K, int lda, int ldb, int ldc,
                         long sA, long sB, long sC, int batch,
                         float alpha, float beta, int flags) {
    dim3 grd((unsigned)(N / 64), (unsigned)(M / 64), (unsigned)batch), blk(128);
    if (bkn)
      gemm_wmma_h<1><<<grd, blk, 0, stream>>>(A, Bm, bias, Cf, Ch, M, N, K, lda, ldb, ldc,
                                              sA, sB, sC, alpha, beta, flags);
    else
      gemm_wmma_h<0><<<grd, blk, 0, stream>>>(A, Bm, bias, Cf, Ch, M, N, K, lda, ldb, ldc,
                                              sA, sB, sC, alpha, beta, flags);
  };

  // ---- one-time f32 -> f16 operand conversion ----
  f32_to_f16<<<gsz(4 * 4096 * 1536), 256, 0, stream>>>(data, data_h, 4 * 4096 * 1536);
  f32_to_f16<<<gsz(512 * 1536), 256, 0, stream>>>(W1, W1_h, 512 * 1536);
  f32_to_f16<<<gsz(1536 * 512), 256, 0, stream>>>(Wqkv1, Wqkv1_h, 1536 * 512);
  f32_to_f16<<<gsz(1536 * 512), 256, 0, stream>>>(Wqkv2, Wqkv2_h, 1536 * 512);
  f32_to_f16<<<gsz(512 * 512), 256, 0, stream>>>(Wo1, Wo1_h, 512 * 512);
  f32_to_f16<<<gsz(512 * 512), 256, 0, stream>>>(Wo2, Wo2_h, 512 * 512);

  // ---- Nystrom attention layer ----
  auto attention = [&](const float* lnG, const float* lnB, const _Float16* Wqkv_h,
                       const _Float16* Wo_h, const float* bo, const float* resw) {
    layernorm_pad<<<4 * 4352, 128, 0, stream>>>(hbuf, lnG, lnB, xn_h);
    // qkv = xn @ Wqkv^T  (17408x1536x512) -> f16
    launch_gemm(false, xn_h, Wqkv_h, nullptr, nullptr, qkv_h, 17408, 1536, 512, 512, 512, 1536,
                0, 0, 0, 1, 1.0f, 0.0f, 0);
    const int tqkv = 32 * 4352 * 64;
    qkv_repack<<<gsz(tqkv), 256, 0, stream>>>(qkv_h, qb, kb, vb, tqkv);
    const int tl = 32 * 256 * 64;
    landmark_mean<<<gsz(tl), 256, 0, stream>>>(qb, kb, qland, kland, tl);
    // s1 = q @ kland^T ; s2 = qland @ kland^T ; s3 = qland @ k^T  (scores -> f16)
    launch_gemm(false, qb, kland, nullptr, nullptr, s1h, 4352, 256, 64, 64, 64, 256,
                4352L * 64, 256L * 64, 4352L * 256, 32, 1.0f, 0.0f, 0);
    launch_gemm(false, qland, kland, nullptr, nullptr, s2h, 256, 256, 64, 64, 64, 256,
                256L * 64, 256L * 64, 256L * 256, 32, 1.0f, 0.0f, 0);
    launch_gemm(false, qland, kb, nullptr, nullptr, s3h, 256, 4352, 64, 64, 64, 4352,
                256L * 64, 4352L * 64, 256L * 4352, 32, 1.0f, 0.0f, 0);
    softmax_rows_h<<<32 * 4352, 256, 0, stream>>>(s1h, 256);
    softmax_rows_h<<<32 * 256, 256, 0, stream>>>(s2h, 256);
    softmax_rows_h<<<32 * 256, 256, 0, stream>>>(s3h, 4352);
    // pinv(a2) via Newton-Schulz (6 iters), all-WMMA chain in f16
    pinv_sums<<<32, 256, 0, stream>>>(s2h, red);
    pinv_finish<<<1, 256, 0, stream>>>(red);
    const int tz = 32 * 256 * 256;
    transpose_scale<<<gsz(tz), 256, 0, stream>>>(z0h, s2h, red + 16384, tz);
    _Float16* zc = z0h; _Float16* zn = z1h;
    for (int it = 0; it < 6; ++it) {
      launch_gemm(true, s2h, zc, nullptr, nullptr, xzh, 256, 256, 256, 256, 256, 256,
                  65536, 65536, 65536, 32, 1.0f, 0.0f, 0);
      ident_minus<<<gsz(tz), 256, 0, stream>>>(t1h, xzh, 7.0f, tz);            // T1 = 7I - XZ
      launch_gemm(true, xzh, t1h, nullptr, nullptr, t2h, 256, 256, 256, 256, 256, 256,
                  65536, 65536, 65536, 32, -1.0f, 15.0f, EF_IDENT);            // T2 = 15I - XZ*T1
      launch_gemm(true, xzh, t2h, nullptr, nullptr, t1h, 256, 256, 256, 256, 256, 256,
                  65536, 65536, 65536, 32, -1.0f, 13.0f, EF_IDENT);            // T3 = 13I - XZ*T2
      launch_gemm(true, zc, t1h, nullptr, nullptr, zn, 256, 256, 256, 256, 256, 256,
                  65536, 65536, 65536, 32, 0.25f, 0.0f, 0);                    // Z = 0.25*Z*T3
      _Float16* tmp = zc; zc = zn; zn = tmp;
    }
    // out = a1 @ (a2_inv @ (a3 @ v)) + res
    launch_gemm(true, s3h, vb, nullptr, nullptr, w1h, 256, 64, 4352, 4352, 64, 64,
                256L * 4352, 4352L * 64, 256L * 64, 32, 1.0f, 0.0f, 0);
    launch_gemm(true, zc, w1h, nullptr, nullptr, w2h, 256, 64, 256, 256, 64, 64,
                65536, 256L * 64, 256L * 64, 32, 1.0f, 0.0f, 0);
    launch_gemm(true, s1h, w2h, nullptr, core, nullptr, 4352, 64, 256, 256, 64, 64,
                4352L * 256, 256L * 64, 4352L * 64, 32, 1.0f, 0.0f, 0);
    const int tc = 32 * 4352 * 64;
    res_conv_add<<<gsz(tc), 256, 0, stream>>>(core, vb, resw, tc);
    const int tm = 4 * 4352 * 512;
    merge_heads<<<gsz(tm), 256, 0, stream>>>(core, merged_h, tm);
    launch_gemm(false, merged_h, Wo_h, bo, attn, nullptr, 17408, 512, 512, 512, 512, 512,
                0, 0, 0, 1, 1.0f, 0.0f, EF_BIAS);
    const int th = 4 * 4097 * 512;
    residual_add<<<gsz(th), 256, 0, stream>>>(hbuf, attn, th);
  };

  // ---- pipeline ----
  // 1) feat0 = relu(data @ W1^T + b1)
  launch_gemm(false, data_h, W1_h, b1, feat0, nullptr, 16384, 512, 1536, 1536, 1536, 512,
              0, 0, 0, 1, 1.0f, 0.0f, EF_BIAS | EF_RELU);
  // 2) h = [cls | feat0]
  const int th0 = 4 * 4097 * 512;
  assemble_h<<<gsz(th0), 256, 0, stream>>>(feat0, cls, hbuf, th0);
  // 3) attention layer 1
  attention(ln1_g, ln1_b, Wqkv1_h, Wo1_h, bo1, res1);
  // 4) fused 7/5/3 depthwise conv + identity on token grid
  dwconv_fused<<<4 * 512, 256, 0, stream>>>(hbuf, pw7, pb7, pw5, pb5, pw3, pb3);
  // 5) attention layer 2
  attention(ln2_g, ln2_b, Wqkv2_h, Wo2_h, bo2, res2);
  // 6) final LN + classifier head -> d_out
  final_head<<<4, 128, 0, stream>>>(hbuf, lnf_g, lnf_b, W2, b2, (float*)d_out);
}
